// GroupedMLP_62508954026513
// MI455X (gfx1250) — matrix operand order repaired and checked by reading, not compiled, not run on hardware
//
#include <hip/hip_runtime.h>

// Problem constants (match reference: E=8, T=2048, H=2048, I=2048)
#define E_ 8
#define T_ 2048
#define H_ 2048
#define I_ 2048

// Tiling
#define BM 128
#define BN 128
#define BK 32
#define LDS_STRIDE 40   // BK + 8 pad: rows are 80B, 16B-aligned, bank-spread
#define NK1 (H_ / BK)
#define NK2 (I_ / BK)

typedef __attribute__((ext_vector_type(16))) __bf16    bf16x16;
typedef __attribute__((ext_vector_type(8)))  float     floatx8;
typedef __attribute__((ext_vector_type(4)))  int       i32x4;
typedef __attribute__((ext_vector_type(8)))  int       i32x8;
typedef __attribute__((ext_vector_type(4)))  unsigned  u32x4;

union FragAB { bf16x16 v; uint4 q[2]; };

#define AS1 __attribute__((address_space(1)))
#define AS3 __attribute__((address_space(3)))

#if defined(__HIP_DEVICE_COMPILE__) && __has_builtin(__builtin_amdgcn_global_load_async_to_lds_b128)
#define USE_ASYNC_LDS 1
#endif

#if defined(__HIP_DEVICE_COMPILE__) && __has_builtin(__builtin_amdgcn_tensor_load_to_lds)
#define USE_TDM 1
#endif

__device__ __forceinline__ unsigned short f32_to_bf16(float f) {
  union { float f; unsigned u; } x; x.f = f;
  unsigned r = x.u + 0x7FFFu + ((x.u >> 16) & 1u);   // round-to-nearest-even
  return (unsigned short)(r >> 16);
}

// Pack two f32 -> packed bf16x2. RNE-round both, then one v_perm_b32 grabs
// both high halves: dst = {hi[31:16], lo[31:16]}.
__device__ __forceinline__ unsigned pack_bf16x2(float lo, float hi) {
#if defined(__HIP_DEVICE_COMPILE__) && __has_builtin(__builtin_amdgcn_cvt_pk_bf16_f32)
  typedef __attribute__((ext_vector_type(2))) __bf16 bf16x2;
  union { bf16x2 v; unsigned u; } c;
  c.v = __builtin_amdgcn_cvt_pk_bf16_f32(lo, hi);
  return c.u;
#else
  union { float f; unsigned u; } a, b; a.f = lo; b.f = hi;
  const unsigned ar = a.u + 0x7FFFu + ((a.u >> 16) & 1u);
  const unsigned br = b.u + 0x7FFFu + ((b.u >> 16) & 1u);
#if defined(__HIP_DEVICE_COMPILE__) && __has_builtin(__builtin_amdgcn_perm)
  return __builtin_amdgcn_perm(br, ar, 0x07060302u);
#else
  return (ar >> 16) | (br & 0xFFFF0000u);
#endif
#endif
}

__device__ __forceinline__ float fast_rcp(float x) {
#if defined(__HIP_DEVICE_COMPILE__) && __has_builtin(__builtin_amdgcn_rcpf)
  return __builtin_amdgcn_rcpf(x);
#else
  return 1.0f / x;
#endif
}

// One b128 lane-copy global -> LDS. Async (ASYNCcnt) on gfx1250, sync fallback.
__device__ __forceinline__ void copy_b128_to_lds(const void* gsrc, void* ldst) {
#ifdef USE_ASYNC_LDS
  __builtin_amdgcn_global_load_async_to_lds_b128(
      (AS1 i32x4*)(unsigned long long)gsrc, (AS3 i32x4*)ldst, 0, 0);
#else
  *(uint4*)ldst = *(const uint4*)gsrc;
#endif
}

__device__ __forceinline__ void async_wait_all() {
#ifdef USE_ASYNC_LDS
#if __has_builtin(__builtin_amdgcn_s_wait_asynccnt)
  __builtin_amdgcn_s_wait_asynccnt(0);
#else
  asm volatile("s_wait_asynccnt 0x0" ::: "memory");
#endif
#endif
}

__device__ __forceinline__ void tensor_wait_all() {
#if defined(__HIP_DEVICE_COMPILE__) && __has_builtin(__builtin_amdgcn_s_wait_tensorcnt)
  __builtin_amdgcn_s_wait_tensorcnt(0);
#else
  asm volatile("s_wait_tensorcnt 0x0" ::: "memory");
#endif
}

#ifdef USE_TDM
// TDM: DMA one 128x32 bf16 tile (row stride I_ elements) from global into LDS,
// writing padded rows (16 DWORDs data + 4 DWORDs pad = 80B = LDS_STRIDE*2).
// D# layout per CDNA5 ISA 8.3/8.4. Issue from ONE wave per block only.
__device__ __forceinline__ void tdm_load_tile_a(const void* gsrc, void* lds_dst) {
  AS3 unsigned short* lp = (AS3 unsigned short*)lds_dst;
  const unsigned lds_off = (unsigned)(unsigned long long)lp;
  const unsigned long long ga = (unsigned long long)gsrc;

  u32x4 g0;
  g0[0] = 1u;                                                 // count=1, user mode
  g0[1] = lds_off;                                            // lds_addr (bytes)
  g0[2] = (unsigned)(ga & 0xFFFFFFFFu);                       // global_addr[31:0]
  g0[3] = (unsigned)((ga >> 32) & 0x1FFFFFFu) | (2u << 30);   // ga[56:32] | type=2

  i32x8 g1;
  g1[0] = (int)((1u << 16)      // data_size = 1 -> 2 bytes
              | (1u << 20)      // pad_enable
              | (3u << 22)      // pad_interval = 3 -> 16 DWORDs (64B row)
              | (3u << 25));    // pad_amount   = 3 -> 4 DWORDs (16B pad)
  g1[1] = (int)((unsigned)I_ << 16);   // tensor_dim0[15:0] (elems)
  g1[2] = (int)((unsigned)T_ << 16);   // tensor_dim0[31:16]=0 | tensor_dim1[15:0]
  g1[3] = (int)((unsigned)BK << 16);   // tensor_dim1[31:16]=0 | tile_dim0=32
  g1[4] = (int)(unsigned)BM;           // tile_dim1=128 | tile_dim2=0
  g1[5] = (int)(unsigned)I_;           // tensor_dim0_stride[31:0]
  g1[6] = 0;                           // stride[47:32] | tensor_dim1_stride lo
  g1[7] = 0;

  const i32x4 gz4 = {0, 0, 0, 0};                  // groups 2/3 unused (2-D)
  const i32x8 gz8 = {0, 0, 0, 0, 0, 0, 0, 0};      // extra group (clang-23 form)
  __builtin_amdgcn_tensor_load_to_lds(g0, g1, gz4, gz4, gz8, 0);
}
#endif

// A fragment: 16x32 bf16, row M = lane&15.
// lanes 0-15 hold K=0..7 (VGPR0-3) and K=16..23 (VGPR4-7); lanes 16-31: K=8..15, K=24..31.
__device__ __forceinline__ bf16x16 load_frag_a(const unsigned short* s, int row0, int lane) {
  FragAB f;
  const int r    = row0 + (lane & 15);
  const int half = (lane >> 4) * 8;
  const unsigned short* p = s + r * LDS_STRIDE + half;
  f.q[0] = *(const uint4*)(p);
  f.q[1] = *(const uint4*)(p + 16);
  return f.v;
}

// B fragment: 32x16 bf16, col N = lane&15 (tile stored [n][k], trans_b).
// lanes 0-15 hold K=0..7 and K=16..23; lanes 16-31: K=8..15, K=24..31.
__device__ __forceinline__ bf16x16 load_frag_b(const unsigned short* s, int col0, int lane) {
  FragAB f;
  const int n     = col0 + (lane & 15);
  const int khalf = (lane >> 4) * 8;
  const unsigned short* p = s + n * LDS_STRIDE + khalf;
  f.q[0] = *(const uint4*)(p);
  f.q[1] = *(const uint4*)(p + 16);
  return f.v;
}

// ---------------------------------------------------------------------------
// Kernel 1: fc1 (x · w1^T) with fused f32->bf16 staging and SwiGLU epilogue.
// Double-buffered LDS software pipeline; 8 waves, each a 32x64 sub-tile with
// dual accumulators (x1 / x2 halves of w1): 16 WMMAs per K-step per wave.
// ---------------------------------------------------------------------------
__global__ __launch_bounds__(256)
void moe_fc1_swiglu_kernel(const float* __restrict__ x,
                           const float* __restrict__ w1,
                           unsigned short* __restrict__ inter) {
  __shared__ unsigned short As [2][BM * LDS_STRIDE];
  __shared__ unsigned short B1s[2][BN * LDS_STRIDE];
  __shared__ unsigned short B2s[2][BN * LDS_STRIDE];

  const int tid  = threadIdx.x;
  const int lane = tid & 31;
  const int wave = tid >> 5;
  const int wm   = (wave >> 1) * 32;
  const int wn   = (wave &  1) * 64;

  const int e  = blockIdx.z;
  const int m0 = blockIdx.x * BM;
  const int n0 = blockIdx.y * BN;

  const float* Ag  = x  + ((size_t)e * T_ + m0) * (size_t)H_;
  const float* B1g = w1 + ((size_t)e * 2 * I_ + n0)      * (size_t)H_;
  const float* B2g = w1 + ((size_t)e * 2 * I_ + I_ + n0) * (size_t)H_;

  const floatx8 vzero = {0.f,0.f,0.f,0.f,0.f,0.f,0.f,0.f};
  floatx8 acc1[2][4], acc2[2][4];
  #pragma unroll
  for (int i = 0; i < 2; ++i)
    #pragma unroll
    for (int j = 0; j < 4; ++j) { acc1[i][j] = vzero; acc2[i][j] = vzero; }

  float4 ra[4], rb1[4], rb2[4];          // in-flight staging registers

  auto load_globals = [&](int k0) {
    #pragma unroll
    for (int i = 0; i < 4; ++i) {
      const int idx = tid + i * 256;     // 0..1023 over 128x(32/4) float4s
      const int r   = idx >> 3;
      const int c   = (idx & 7) << 2;
      ra [i] = *(const float4*)(Ag  + (size_t)r * H_ + k0 + c);
      rb1[i] = *(const float4*)(B1g + (size_t)r * H_ + k0 + c);
      rb2[i] = *(const float4*)(B2g + (size_t)r * H_ + k0 + c);
    }
  };
  auto store_lds = [&](int buf) {
    #pragma unroll
    for (int i = 0; i < 4; ++i) {
      const int idx = tid + i * 256;
      const int r   = idx >> 3;
      const int c   = (idx & 7) << 2;
      uint2 pa, pb1, pb2;
      pa.x  = pack_bf16x2(ra [i].x, ra [i].y);
      pa.y  = pack_bf16x2(ra [i].z, ra [i].w);
      pb1.x = pack_bf16x2(rb1[i].x, rb1[i].y);
      pb1.y = pack_bf16x2(rb1[i].z, rb1[i].w);
      pb2.x = pack_bf16x2(rb2[i].x, rb2[i].y);
      pb2.y = pack_bf16x2(rb2[i].z, rb2[i].w);
      *(uint2*)(&As [buf][r * LDS_STRIDE + c]) = pa;
      *(uint2*)(&B1s[buf][r * LDS_STRIDE + c]) = pb1;
      *(uint2*)(&B2s[buf][r * LDS_STRIDE + c]) = pb2;
    }
  };

  // Pipeline prologue
  load_globals(0);
  store_lds(0);
  __syncthreads();

  for (int kt = 0; kt < NK1; ++kt) {
    const int  cur  = kt & 1;
    const int  nxt  = cur ^ 1;
    const bool more = (kt + 1) < NK1;

    if (more) load_globals((kt + 1) * BK);   // prefetch: overlaps with WMMAs

    bf16x16 af[2], bf1[4], bf2[4];
    #pragma unroll
    for (int mt = 0; mt < 2; ++mt) af[mt] = load_frag_a(As[cur], wm + mt * 16, lane);
    #pragma unroll
    for (int nt = 0; nt < 4; ++nt) {
      bf1[nt] = load_frag_b(B1s[cur], wn + nt * 16, lane);
      bf2[nt] = load_frag_b(B2s[cur], wn + nt * 16, lane);
    }
    #pragma unroll
    for (int mt = 0; mt < 2; ++mt)
      #pragma unroll
      for (int nt = 0; nt < 4; ++nt) {
        acc1[mt][nt] = __builtin_amdgcn_wmma_f32_16x16x32_bf16(
            false, af[mt], false, bf1[nt], (short)0, acc1[mt][nt], false, false);
        acc2[mt][nt] = __builtin_amdgcn_wmma_f32_16x16x32_bf16(
            false, af[mt], false, bf2[nt], (short)0, acc2[mt][nt], false, false);
      }

    if (more) store_lds(nxt);                // writes other buffer: no hazard
    __syncthreads();
  }

  // Epilogue: inter = x1 * silu(x2). C layout: VGPR r -> M = r + 8*(lane>>4), N = lane&15
  const int col  = lane & 15;
  const int rofs = (lane >> 4) * 8;
  unsigned short* Og = inter + (size_t)e * T_ * I_;
  #pragma unroll
  for (int mt = 0; mt < 2; ++mt)
    #pragma unroll
    for (int nt = 0; nt < 4; ++nt)
      #pragma unroll
      for (int r = 0; r < 8; ++r) {
        const float a1  = acc1[mt][nt][r];
        const float a2  = acc2[mt][nt][r];
        const float sil = a2 * fast_rcp(1.0f + __expf(-a2));
        const int row = m0 + wm + mt * 16 + rofs + r;
        const int cc  = n0 + wn + nt * 16 + col;
        Og[(size_t)row * I_ + cc] = f32_to_bf16(a1 * sil);
      }
}

// ---------------------------------------------------------------------------
// Kernel 2: fc2 (inter · w2^T). A (bf16) is staged by the Tensor Data Mover:
// one D#-driven DMA per 128x32 tile, LDS-padded rows, TENSORcnt-tracked
// (fallback: per-lane async global->LDS b128). B is f32->bf16 via registers.
// ---------------------------------------------------------------------------
__global__ __launch_bounds__(256)
void moe_fc2_kernel(const unsigned short* __restrict__ inter,
                    const float* __restrict__ w2,
                    unsigned short* __restrict__ out) {
  __shared__ unsigned short As[2][BM * LDS_STRIDE];
  __shared__ unsigned short Bs[2][BN * LDS_STRIDE];

  const int tid  = threadIdx.x;
  const int lane = tid & 31;
  const int wave = tid >> 5;
  const int wm   = (wave >> 1) * 32;
  const int wn   = (wave &  1) * 64;

  const int e  = blockIdx.z;
  const int m0 = blockIdx.x * BM;
  const int n0 = blockIdx.y * BN;

  const unsigned short* Ag = inter + ((size_t)e * T_ + m0) * (size_t)I_;
  const float*          Bg = w2    + ((size_t)e * H_ + n0) * (size_t)I_;

  const floatx8 vzero = {0.f,0.f,0.f,0.f,0.f,0.f,0.f,0.f};
  floatx8 acc[2][4];
  #pragma unroll
  for (int i = 0; i < 2; ++i)
    #pragma unroll
    for (int j = 0; j < 4; ++j) acc[i][j] = vzero;

  float4 rb[4];

  auto stage_a = [&](int k0, int buf) {
#ifdef USE_TDM
    if (wave == 0) tdm_load_tile_a(Ag + k0, &As[buf][0]);   // one DMA per block
#else
    #pragma unroll
    for (int i = 0; i < 2; ++i) {
      const int idx = tid + i * 256;      // 0..511 over 128x(32/8) uint4s
      const int r   = idx >> 2;
      const int c   = (idx & 3) << 3;
      copy_b128_to_lds(Ag + (size_t)r * I_ + k0 + c, &As[buf][r * LDS_STRIDE + c]);
    }
#endif
  };
  auto stage_a_wait = [&]() {
#ifdef USE_TDM
    if (wave == 0) tensor_wait_all();
#else
    async_wait_all();
#endif
  };
  auto load_b = [&](int k0) {
    #pragma unroll
    for (int i = 0; i < 4; ++i) {
      const int idx = tid + i * 256;
      const int r   = idx >> 3;
      const int c   = (idx & 7) << 2;
      rb[i] = *(const float4*)(Bg + (size_t)r * I_ + k0 + c);
    }
  };
  auto store_b = [&](int buf) {
    #pragma unroll
    for (int i = 0; i < 4; ++i) {
      const int idx = tid + i * 256;
      const int r   = idx >> 3;
      const int c   = (idx & 7) << 2;
      uint2 pb;
      pb.x = pack_bf16x2(rb[i].x, rb[i].y);
      pb.y = pack_bf16x2(rb[i].z, rb[i].w);
      *(uint2*)(&Bs[buf][r * LDS_STRIDE + c]) = pb;
    }
  };

  // Pipeline prologue
  stage_a(0, 0);
  load_b(0);
  store_b(0);
  stage_a_wait();
  __syncthreads();

  for (int kt = 0; kt < NK2; ++kt) {
    const int  cur  = kt & 1;
    const int  nxt  = cur ^ 1;
    const bool more = (kt + 1) < NK2;

    if (more) {
      stage_a((kt + 1) * BK, nxt);        // TDM DMA into other buffer
      load_b((kt + 1) * BK);              // prefetch B into registers
    }

    bf16x16 af[2], bfr[4];
    #pragma unroll
    for (int mt = 0; mt < 2; ++mt) af[mt] = load_frag_a(As[cur], wm + mt * 16, lane);
    #pragma unroll
    for (int nt = 0; nt < 4; ++nt) bfr[nt] = load_frag_b(Bs[cur], wn + nt * 16, lane);
    #pragma unroll
    for (int mt = 0; mt < 2; ++mt)
      #pragma unroll
      for (int nt = 0; nt < 4; ++nt)
        acc[mt][nt] = __builtin_amdgcn_wmma_f32_16x16x32_bf16(
            false, af[mt], false, bfr[nt], (short)0, acc[mt][nt], false, false);

    if (more) store_b(nxt);
    stage_a_wait();                        // DMA into nxt complete pre-barrier
    __syncthreads();
  }

  const int col  = lane & 15;
  const int rofs = (lane >> 4) * 8;
  unsigned short* Og = out + (size_t)e * T_ * H_;
  #pragma unroll
  for (int mt = 0; mt < 2; ++mt)
    #pragma unroll
    for (int nt = 0; nt < 4; ++nt)
      #pragma unroll
      for (int r = 0; r < 8; ++r) {
        const int row = m0 + wm + mt * 16 + rofs + r;
        const int cc  = n0 + wn + nt * 16 + col;
        Og[(size_t)row * H_ + cc] = f32_to_bf16(acc[mt][nt][r]);
      }
}

extern "C" void kernel_launch(void* const* d_in, const int* in_sizes, int n_in,
                              void* d_out, int out_size, void* d_ws, size_t ws_size,
                              hipStream_t stream) {
  (void)in_sizes; (void)n_in; (void)out_size; (void)ws_size;
  const float* x  = (const float*)d_in[0];
  // d_in[1] = tokens_per_expert (uniform T_PER_E, unused)
  const float* w1 = (const float*)d_in[2];
  const float* w2 = (const float*)d_in[3];
  unsigned short* inter = (unsigned short*)d_ws;   // E*T*I bf16 = 64 MB scratch
  unsigned short* outp  = (unsigned short*)d_out;  // bf16 output

  dim3 block(256);
  dim3 g1(T_ / BM, I_ / BN, E_);
  moe_fc1_swiglu_kernel<<<g1, block, 0, stream>>>(x, w1, inter);
  dim3 g2(T_ / BM, H_ / BN, E_);
  moe_fc2_kernel<<<g2, block, 0, stream>>>(inter, w2, outp);
}
